// Llama4Experts_64244120814198
// MI455X (gfx1250) — matrix operand order repaired and checked by reading, not compiled, not run on hardware
//
#include <hip/hip_runtime.h>
#include <hip/hip_bf16.h>

// Llama4-style per-expert MLP: (E,T,H)@(E,H,2I) -> SwiGLU -> @(E,I,H)
// E=16, T=64, H=2048, I=4096. fp32 in/out, bf16 WMMA compute (f32 accum).
// A-tiles staged via TDM (tensor_load_to_lds) with ping-pong LDS buffers;
// B-tiles register-prefetched, converted f32->bf16x2, stored transposed.

static constexpr int NE = 16;
static constexpr int NT = 64;
static constexpr int NH = 2048;
static constexpr int NI = 4096;

// Padded LDS k-stride: 40 ushorts = 80 bytes (16B-aligned for ds_load_b128).
static constexpr int LDSK = 40;

typedef __attribute__((ext_vector_type(16))) __bf16 v16bf;
typedef __attribute__((ext_vector_type(8)))  float  v8f;
typedef unsigned int u32x4 __attribute__((ext_vector_type(4)));
typedef int          i32x4 __attribute__((ext_vector_type(4)));
typedef int          i32x8 __attribute__((ext_vector_type(8)));

union Frag {
    v16bf v;
    uint4 q[2];
};

__device__ __forceinline__ unsigned short f32_bf16_rne(float f) {
    unsigned int u = __float_as_uint(f);
    u += 0x7FFFu + ((u >> 16) & 1u);
    return (unsigned short)(u >> 16);
}

__device__ __forceinline__ unsigned pack_bf16x2(float lo, float hi) {
#if __has_builtin(__builtin_amdgcn_cvt_pk_bf16_f32)
    typedef __bf16 bf16x2 __attribute__((ext_vector_type(2)));
    union { bf16x2 v; unsigned u; } r;
    r.v = __builtin_amdgcn_cvt_pk_bf16_f32(lo, hi);
    return r.u;
#else
    unsigned ul = __float_as_uint(lo);
    ul += 0x7FFFu + ((ul >> 16) & 1u);
    unsigned uh = __float_as_uint(hi);
    uh += 0x7FFFu + ((uh >> 16) & 1u);
    return (ul >> 16) | (uh & 0xFFFF0000u);
#endif
}

__device__ __forceinline__ float f4c(const float4& v, int j) {
    return j == 0 ? v.x : j == 1 ? v.y : j == 2 ? v.z : v.w;
}

// ---------------------------------------------------------------------------
// TDM: load a 2D bf16 tile (tile_d0 elems along contiguous dim, tile_d1 rows)
// from global to LDS, with LDS padding 16B per 64B row -> 80B row stride.
// ---------------------------------------------------------------------------
__device__ __forceinline__ void tdm_load_tile_bf16(
    unsigned lds_off, unsigned long long gaddr,
    unsigned td0, unsigned td1,            // tensor dims (elements)
    unsigned tile_d0, unsigned tile_d1,    // tile dims (elements)
    unsigned long long stride0)            // row stride (elements)
{
    u32x4 g0;
    g0[0] = 1u;                                            // count=1, user D#
    g0[1] = lds_off;                                       // LDS byte address
    g0[2] = (unsigned)(gaddr & 0xFFFFFFFFull);             // global addr lo
    g0[3] = (unsigned)((gaddr >> 32) & 0x01FFFFFFull)      // global addr hi
            | (2u << 30);                                  // type = 2 (image)
    i32x8 g1;
    // data_size=1 (2B), pad_enable, pad_interval=3 (64B), pad_amount=3 (16B)
    g1[0] = (int)((1u << 16) | (1u << 20) | (3u << 22) | (3u << 25));
    g1[1] = (int)((td0 & 0xFFFFu) << 16);                  // tensor_dim0 lo
    g1[2] = (int)((td0 >> 16) | ((td1 & 0xFFFFu) << 16));  // dim0 hi, dim1 lo
    g1[3] = (int)((td1 >> 16) | (tile_d0 << 16));          // dim1 hi, tile_dim0
    g1[4] = (int)(tile_d1 & 0xFFFFu);                      // tile_dim1 (dim2=0)
    g1[5] = (int)(unsigned)(stride0 & 0xFFFFFFFFull);      // dim0_stride lo
    g1[6] = (int)(unsigned)((stride0 >> 32) & 0xFFFFull);  // dim0_stride hi
    g1[7] = 0;
    i32x4 z4 = {0, 0, 0, 0};
#if defined(__clang_major__) && (__clang_major__ >= 23)
    i32x8 z8 = {0, 0, 0, 0, 0, 0, 0, 0};
    __builtin_amdgcn_tensor_load_to_lds(g0, g1, z4, z4, z8, 0);
#else
    __builtin_amdgcn_tensor_load_to_lds(g0, g1, z4, z4, 0);
#endif
}

// ---------------------------------------------------------------------------
// Kernel 1: hidden_states f32 -> bf16 (8 elements / thread)
// ---------------------------------------------------------------------------
__global__ __launch_bounds__(256) void cvt_bf16_kernel(
    const float* __restrict__ x, unsigned short* __restrict__ y, int n8) {
    int i = blockIdx.x * blockDim.x + threadIdx.x;
    if (i >= n8) return;
    const float4* p = (const float4*)x + (size_t)i * 2;
    float4 a = p[0];
    float4 b = p[1];
    uint4 o;
    o.x = pack_bf16x2(a.x, a.y);
    o.y = pack_bf16x2(a.z, a.w);
    o.z = pack_bf16x2(b.x, b.y);
    o.w = pack_bf16x2(b.z, b.w);
    ((uint4*)y)[i] = o;
}

// ---------------------------------------------------------------------------
// Kernel 2: gate_up GEMM + fused SwiGLU -> bf16 act.
// grid = (E, NI/64); block = 256 threads (8 waves).
// Wave w: rows 16*(w&3).., cols 32*(w>>2).. of this block's 64-col act slab;
// each wave holds gate AND up accumulators -> in-register SwiGLU.
// ---------------------------------------------------------------------------
__global__ __launch_bounds__(256) void gateup_swiglu_kernel(
    const unsigned short* __restrict__ Xb,   // (E,T,H) bf16
    const float* __restrict__ W,             // (E,H,2I) f32
    unsigned short* __restrict__ act)        // (E,T,I) bf16
{
    __shared__ unsigned short sA[2][NT][LDSK];  // ping-pong 64x32 bf16 (TDM)
    __shared__ unsigned short sB[128][LDSK];    // [n][k]; n<64 gate, n>=64 up

    const int e    = blockIdx.x;
    const int nb   = blockIdx.y;                // act cols nb*64 .. +63
    const int tid  = threadIdx.x;
    const int wave = tid >> 5;
    const int lane = tid & 31;
    const int mtile = wave & 3;
    const int nhalf = wave >> 2;
    const int fr = lane & 15;
    const int hi = lane >> 4;

    const unsigned short* Xe = Xb + (size_t)e * NT * NH;
    const float*          We = W  + (size_t)e * NH * (2 * NI);

    // B staging: thread owns a 2(k) x 8(n) micro-tile.
    const int kp   = tid >> 4;                  // k-pair 0..15 -> rows 2kp,2kp+1
    const int nc8  = (tid & 15) * 8;            // 0..120
    const int gcol = (nc8 < 64) ? (nb * 64 + nc8)
                                : (NI + nb * 64 + (nc8 - 64));

    const unsigned sA0 = (unsigned)(uintptr_t)&sA[0][0][0];
    const unsigned sA1 = (unsigned)(uintptr_t)&sA[1][0][0];

    v8f ag0 = {}, ag1 = {}, au0 = {}, au1 = {};

    // ---- prologue: TDM A(k=0) -> sA[0]; prefetch B(k=0) into regs ----
    if (wave == 0) {
        tdm_load_tile_bf16(sA0, (unsigned long long)(uintptr_t)Xe,
                           NH, NT, 32, NT, NH);
        __builtin_amdgcn_s_wait_tensorcnt(0);
    }
    const float* Bp = We + (size_t)(2 * kp) * (2 * NI) + gcol;
    float4 p0 = *(const float4*)(Bp);
    float4 p1 = *(const float4*)(Bp + 4);
    float4 p2 = *(const float4*)(Bp + 2 * NI);
    float4 p3 = *(const float4*)(Bp + 2 * NI + 4);
    __syncthreads();

    int p = 0;
    for (int k0 = 0; k0 < NH; k0 += 32, p ^= 1) {
        const int kn = (k0 + 32 < NH) ? (k0 + 32) : 0;  // wrap: no OOB tail

        // TDM prefetch of next A tile into the other buffer
        if (wave == 0) {
            tdm_load_tile_bf16(p ? sA0 : sA1,
                               (unsigned long long)(uintptr_t)(Xe + kn),
                               NH - kn, NT, 32, NT, NH);
        }

        // convert prefetched B regs -> LDS, packed bf16x2 along k (b32 stores)
#pragma unroll
        for (int j = 0; j < 4; ++j) {
            *(unsigned*)&sB[nc8 + j][2 * kp]     = pack_bf16x2(f4c(p0, j), f4c(p2, j));
            *(unsigned*)&sB[nc8 + 4 + j][2 * kp] = pack_bf16x2(f4c(p1, j), f4c(p3, j));
        }
        __syncthreads();                        // sB ready (sA[p] ready earlier)

        // register-prefetch next B tile (latency hidden under WMMA section)
        const float* Bn = We + (size_t)(kn + 2 * kp) * (2 * NI) + gcol;
        p0 = *(const float4*)(Bn);
        p1 = *(const float4*)(Bn + 4);
        p2 = *(const float4*)(Bn + 2 * NI);
        p3 = *(const float4*)(Bn + 2 * NI + 4);

        // ---- fragments (k = 16*hi + s in A and B) + WMMA ----
        Frag a, bg0, bg1, bu0, bu1;
        const unsigned short* ap = &sA[p][mtile * 16 + fr][0];
        a.q[0] = *(const uint4*)(ap + hi * 16);         // K[hi*16..+7]
        a.q[1] = *(const uint4*)(ap + hi * 16 + 8);     // K[hi*16+8..+7]
        const int cg = nhalf * 32 + fr;
        const unsigned short* bp0 = &sB[cg][hi * 16];
        const unsigned short* bp1 = &sB[cg + 16][hi * 16];
        const unsigned short* bp2 = &sB[64 + cg][hi * 16];
        const unsigned short* bp3 = &sB[64 + cg + 16][hi * 16];
        bg0.q[0] = *(const uint4*)(bp0); bg0.q[1] = *(const uint4*)(bp0 + 8);
        bg1.q[0] = *(const uint4*)(bp1); bg1.q[1] = *(const uint4*)(bp1 + 8);
        bu0.q[0] = *(const uint4*)(bp2); bu0.q[1] = *(const uint4*)(bp2 + 8);
        bu1.q[0] = *(const uint4*)(bp3); bu1.q[1] = *(const uint4*)(bp3 + 8);

        ag0 = __builtin_amdgcn_wmma_f32_16x16x32_bf16(false, a.v, false, bg0.v,
                                                      (short)0, ag0, false, false);
        ag1 = __builtin_amdgcn_wmma_f32_16x16x32_bf16(false, a.v, false, bg1.v,
                                                      (short)0, ag1, false, false);
        au0 = __builtin_amdgcn_wmma_f32_16x16x32_bf16(false, a.v, false, bu0.v,
                                                      (short)0, au0, false, false);
        au1 = __builtin_amdgcn_wmma_f32_16x16x32_bf16(false, a.v, false, bu1.v,
                                                      (short)0, au1, false, false);

        if (wave == 0) __builtin_amdgcn_s_wait_tensorcnt(0);
        __syncthreads();                        // sB consumed; sA[p^1] ready
    }

    // ---- fused SwiGLU: act = up * gate * rcp(1+exp(-gate)), store bf16 ----
    const int mbase   = mtile * 16 + hi * 8;    // C/D: M = mbase + r, N = fr
    const int colbase = nb * 64 + nhalf * 32;
#pragma unroll
    for (int r = 0; r < 8; ++r) {
        size_t rowoff = ((size_t)e * NT + mbase + r) * NI;
        {
            float g = ag0[r], u = au0[r];
            float s = u * g * __builtin_amdgcn_rcpf(1.0f + __expf(-g));
            act[rowoff + colbase + fr] = f32_bf16_rne(s);
        }
        {
            float g = ag1[r], u = au1[r];
            float s = u * g * __builtin_amdgcn_rcpf(1.0f + __expf(-g));
            act[rowoff + colbase + 16 + fr] = f32_bf16_rne(s);
        }
    }
}

// ---------------------------------------------------------------------------
// Kernel 3: down projection. grid = (E, NH/64). f32 output.
// ---------------------------------------------------------------------------
__global__ __launch_bounds__(256) void down_kernel(
    const unsigned short* __restrict__ act,  // (E,T,I) bf16
    const float* __restrict__ Wd,            // (E,I,H) f32
    float* __restrict__ out)                 // (E,T,H) f32
{
    __shared__ unsigned short sA[2][NT][LDSK];
    __shared__ unsigned short sB[64][LDSK];

    const int e    = blockIdx.x;
    const int nb   = blockIdx.y;                // out cols nb*64 .. +63
    const int tid  = threadIdx.x;
    const int wave = tid >> 5;
    const int lane = tid & 31;
    const int mtile = wave & 3;
    const int nhalf = wave >> 2;
    const int fr = lane & 15;
    const int hi = lane >> 4;

    const unsigned short* Ae = act + (size_t)e * NT * NI;
    const float*          We = Wd  + (size_t)e * NI * NH;

    // B staging: thread owns a 2(k) x 4(n) micro-tile.
    const int kp  = tid >> 4;                   // 0..15 -> rows 2kp, 2kp+1
    const int nc4 = (tid & 15) * 4;             // 0..60
    const int bcol = nb * 64 + nc4;

    const unsigned sA0 = (unsigned)(uintptr_t)&sA[0][0][0];
    const unsigned sA1 = (unsigned)(uintptr_t)&sA[1][0][0];

    v8f c0 = {}, c1 = {};

    if (wave == 0) {
        tdm_load_tile_bf16(sA0, (unsigned long long)(uintptr_t)Ae,
                           NI, NT, 32, NT, NI);
        __builtin_amdgcn_s_wait_tensorcnt(0);
    }
    const float* Bp = We + (size_t)(2 * kp) * NH + bcol;
    float4 p0 = *(const float4*)(Bp);
    float4 p1 = *(const float4*)(Bp + NH);
    __syncthreads();

    int p = 0;
    for (int k0 = 0; k0 < NI; k0 += 32, p ^= 1) {
        const int kn = (k0 + 32 < NI) ? (k0 + 32) : 0;

        if (wave == 0) {
            tdm_load_tile_bf16(p ? sA0 : sA1,
                               (unsigned long long)(uintptr_t)(Ae + kn),
                               NI - kn, NT, 32, NT, NI);
        }

#pragma unroll
        for (int j = 0; j < 4; ++j) {
            *(unsigned*)&sB[nc4 + j][2 * kp] = pack_bf16x2(f4c(p0, j), f4c(p1, j));
        }
        __syncthreads();

        const float* Bn = We + (size_t)(kn + 2 * kp) * NH + bcol;
        p0 = *(const float4*)(Bn);
        p1 = *(const float4*)(Bn + NH);

        Frag a, b0, b1;
        const unsigned short* ap = &sA[p][mtile * 16 + fr][0];
        a.q[0] = *(const uint4*)(ap + hi * 16);
        a.q[1] = *(const uint4*)(ap + hi * 16 + 8);
        const int cg = nhalf * 32 + fr;
        const unsigned short* bp0 = &sB[cg][hi * 16];
        const unsigned short* bp1 = &sB[cg + 16][hi * 16];
        b0.q[0] = *(const uint4*)(bp0); b0.q[1] = *(const uint4*)(bp0 + 8);
        b1.q[0] = *(const uint4*)(bp1); b1.q[1] = *(const uint4*)(bp1 + 8);

        c0 = __builtin_amdgcn_wmma_f32_16x16x32_bf16(false, a.v, false, b0.v,
                                                     (short)0, c0, false, false);
        c1 = __builtin_amdgcn_wmma_f32_16x16x32_bf16(false, a.v, false, b1.v,
                                                     (short)0, c1, false, false);

        if (wave == 0) __builtin_amdgcn_s_wait_tensorcnt(0);
        __syncthreads();
    }

    const int mbase   = mtile * 16 + hi * 8;
    const int colbase = nb * 64 + nhalf * 32;
#pragma unroll
    for (int r = 0; r < 8; ++r) {
        size_t rowoff = ((size_t)e * NT + mbase + r) * NH;
        out[rowoff + colbase + fr]      = c0[r];
        out[rowoff + colbase + 16 + fr] = c1[r];
    }
}

// ---------------------------------------------------------------------------
extern "C" void kernel_launch(void* const* d_in, const int* in_sizes, int n_in,
                              void* d_out, int out_size, void* d_ws, size_t ws_size,
                              hipStream_t stream) {
    const float* hs  = (const float*)d_in[0];   // (E,T,H)
    const float* gup = (const float*)d_in[1];   // (E,H,2I)
    const float* dwn = (const float*)d_in[2];   // (E,I,H)
    float* out = (float*)d_out;                 // (E,T,H)

    unsigned short* hbf  = (unsigned short*)d_ws;           // E*T*H bf16
    unsigned short* actb = hbf + (size_t)NE * NT * NH;      // E*T*I bf16

    int n8 = (NE * NT * NH) / 8;
    cvt_bf16_kernel<<<(n8 + 255) / 256, 256, 0, stream>>>(hs, hbf, n8);

    gateup_swiglu_kernel<<<dim3(NE, NI / 64), 256, 0, stream>>>(hbf, gup, actb);

    down_kernel<<<dim3(NE, NH / 64), 256, 0, stream>>>(actb, dwn, out);
}
